// FeedForwardQuantum_65481071407856
// MI455X (gfx1250) — matrix operand order repaired and checked by reading, not compiled, not run on hardware
//
#include <hip/hip_runtime.h>
#include <hip/hip_bf16.h>

// ---------------------------------------------------------------------------
// Problem constants (fixed by the reference)
// ---------------------------------------------------------------------------
#define NQ        10
#define EMBED     512
#define FFN       2048
#define NTOK      16384                 // 16 * 1024 tokens

typedef __attribute__((ext_vector_type(16))) _Float16 v16h;
typedef __attribute__((ext_vector_type(8)))  _Float16 v8h;
typedef __attribute__((ext_vector_type(8)))  float    v8f;
typedef __attribute__((ext_vector_type(4)))  int      v4i;

// Pointer types matching __builtin_amdgcn_global_load_async_to_lds_b128:
//   (v4i addrspace(1)*, v4i addrspace(3)*, int offset, int cpol)
typedef __attribute__((address_space(1))) v4i gm_v4i;
typedef __attribute__((address_space(3))) v4i lds_v4i;

#if defined(__has_builtin)
#  if __has_builtin(__builtin_amdgcn_global_load_async_to_lds_b128)
#    define HAS_ASYNC_LDS 1
#  endif
#endif

static __device__ __forceinline__ void wait_async0() {
#if defined(__has_builtin) && __has_builtin(__builtin_amdgcn_s_wait_asynccnt)
    __builtin_amdgcn_s_wait_asynccnt(0);
#else
    asm volatile("s_wait_asynccnt 0x0" ::: "memory");
#endif
}

// ---------------------------------------------------------------------------
// Kernel 0: w2 (512x2048 f32, row-major) -> f16, same layout.
// Row-major (N,K): each WMMA B-fragment column reads contiguous K halfs.
// ---------------------------------------------------------------------------
__global__ __launch_bounds__(256) void w2_cvt_kernel(const float* __restrict__ w2,
                                                     _Float16* __restrict__ w2h) {
    int i = blockIdx.x * 256 + threadIdx.x;      // grid sized exactly
    w2h[i] = (_Float16)w2[i];
}

// ---------------------------------------------------------------------------
// Kernel 1: fused quantum layer + GEMM1 (+bias, relu), h stored f16 row-major.
//   Closed form of the 10-qubit circuit (RX compose; CNOT chain = prefix XOR;
//   product state => expectations are products of cos):
//     q[0]   = prod_{w=1..9} cos(x_w + rxp_w)
//     q[k>0] = prod_{w=0..k} cos(x_w + rxp_w)
//   h[t,c] = relu(b1[c] + sum_j q[j] * w1[c,j])      (K=10 -> VALU, f32 exact)
// ---------------------------------------------------------------------------
__global__ __launch_bounds__(256) void qffn1_kernel(const float* __restrict__ x,
                                                    const float* __restrict__ rxp,
                                                    const float* __restrict__ w1,
                                                    const float* __restrict__ b1,
                                                    _Float16* __restrict__ h) {
    __shared__ float qs[NQ];
    const int token = blockIdx.x;

    if (threadIdx.x == 0) {
        float c[NQ];
#pragma unroll
        for (int w = 0; w < NQ; ++w)
            c[w] = cosf(x[(size_t)token * EMBED + w] + rxp[w]);
        float p = 1.0f;                       // q[0] = prod c[1..9]
#pragma unroll
        for (int w = 1; w < NQ; ++w) p *= c[w];
        qs[0] = p;
        p = c[0];                             // q[k] = prod c[0..k]
#pragma unroll
        for (int k = 1; k < NQ; ++k) { p *= c[k]; qs[k] = p; }
    }
    __syncthreads();

    float q[NQ];
#pragma unroll
    for (int j = 0; j < NQ; ++j) q[j] = qs[j];

#pragma unroll
    for (int i = 0; i < FFN / 256; ++i) {
        const int col = threadIdx.x + i * 256;
        float acc = b1[col];
#pragma unroll
        for (int j = 0; j < NQ; ++j)
            acc = fmaf(q[j], w1[col * NQ + j], acc);
        acc = fmaxf(acc, 0.0f);
        h[(size_t)token * FFN + col] = (_Float16)acc;
    }
}

// ---------------------------------------------------------------------------
// Kernel 2: out = h(16384x2048 f16) @ w2^T + b2 via v_wmma_f32_16x16x32_f16.
//
// Block = 256 threads = 8 waves; block tile 128M x 128N, K-steps of 32.
// Waves: mg = wave&3 (2 M-tiles each), ng = wave>>2 (4 N-tiles each)
//   -> 8 WMMA / wave / K-step, A frag reused x4, B frag reused x2.
// A and B K-chunks are double-buffered in LDS (async global->LDS when the
// toolchain exposes the builtin; register staging otherwise).  LDS rows are
// padded to 40 halfs so per-lane ds_load_b128 fragment reads are conflict-free.
// The K-loop is kept at unroll 1: the body already overlaps (async staging of
// step s+1 runs under the WMMAs of step s), and unrolling was observed to
// force accumulator v_mov shuffles + WMMA->VALU v_nop hazards.
//
// Fragment layouts per CDNA5 ISA 7.12.2 (wave32):
//   A (16x32 f16): lane -> M = lane&15; halfs 0..7  = K + 8*(lane>>4),
//                                        halfs 8..15 = K + 16 + 8*(lane>>4)
//   B (32x16 f16): lane -> N = lane&15; halfs 0..7  = K + 8*(lane>>4),
//                                        halfs 8..15 = K + 16 + 8*(lane>>4)
//   D (16x16 f32): vgpr v -> M = v + 8*(lane>>4), N = lane&15
// ---------------------------------------------------------------------------
#define BLK_M   128
#define BLK_N   128
#define LDSTR   40                       // padded halfs per 32-half K-row
#define NSTEP   (FFN / 32)               // 64 K-steps

__global__ __launch_bounds__(256) void qffn2_wmma_kernel(const _Float16* __restrict__ h,
                                                         const _Float16* __restrict__ w2h,
                                                         const float* __restrict__ b2,
                                                         float* __restrict__ out) {
    __shared__ _Float16 lsA[2][BLK_M * LDSTR];   // 10 KB per buffer
    __shared__ _Float16 lsB[2][BLK_N * LDSTR];

    const int tid   = threadIdx.x;
    const int lane  = tid & 31;
    const int wave  = tid >> 5;
    const int l16   = lane & 15;
    const int lhalf = lane >> 4;                 // 0 | 1
    const int mg    = wave & 3;                  // 4 M-groups of 32 rows
    const int ng    = wave >> 2;                 // 2 N-groups of 64 cols

    const int mBase = blockIdx.y * BLK_M;
    const int nBase = blockIdx.x * BLK_N;

    // Staging: 2 threads per row/col, 32 B (16 halfs) each, 2 x b128.
    const int srow  = tid & 127;
    const int spart = tid >> 7;                  // halfs [0,16) | [16,32)
    const _Float16* gA = h   + (size_t)(mBase + srow) * FFN + spart * 16;
    const _Float16* gB = w2h + (size_t)(nBase + srow) * FFN + spart * 16;
    const int sdst = srow * LDSTR + spart * 16;

    v8f acc[2][4] = {};

#if HAS_ASYNC_LDS
    #define STAGE_ISSUE(buf, k0)                                                   \
        do {                                                                       \
            __builtin_amdgcn_global_load_async_to_lds_b128(                        \
                (gm_v4i*)(unsigned long long)(gA + (k0)),                          \
                (lds_v4i*)(unsigned)(unsigned long long)&lsA[buf][sdst], 0, 0);    \
            __builtin_amdgcn_global_load_async_to_lds_b128(                        \
                (gm_v4i*)(unsigned long long)(gA + (k0) + 8),                      \
                (lds_v4i*)(unsigned)(unsigned long long)&lsA[buf][sdst + 8], 0, 0);\
            __builtin_amdgcn_global_load_async_to_lds_b128(                        \
                (gm_v4i*)(unsigned long long)(gB + (k0)),                          \
                (lds_v4i*)(unsigned)(unsigned long long)&lsB[buf][sdst], 0, 0);    \
            __builtin_amdgcn_global_load_async_to_lds_b128(                        \
                (gm_v4i*)(unsigned long long)(gB + (k0) + 8),                      \
                (lds_v4i*)(unsigned)(unsigned long long)&lsB[buf][sdst + 8], 0, 0);\
        } while (0)
    #define STAGE_COMMIT(buf) do { } while (0)
    #define STAGE_DRAIN()     wait_async0()
#else
    v8h ra0, ra1, rb0, rb1;
    #define STAGE_ISSUE(buf, k0)                                                   \
        do {                                                                       \
            ra0 = *(const v8h*)(gA + (k0));                                        \
            ra1 = *(const v8h*)(gA + (k0) + 8);                                    \
            rb0 = *(const v8h*)(gB + (k0));                                        \
            rb1 = *(const v8h*)(gB + (k0) + 8);                                    \
        } while (0)
    #define STAGE_COMMIT(buf)                                                      \
        do {                                                                       \
            *(v8h*)&lsA[buf][sdst]     = ra0;                                      \
            *(v8h*)&lsA[buf][sdst + 8] = ra1;                                      \
            *(v8h*)&lsB[buf][sdst]     = rb0;                                      \
            *(v8h*)&lsB[buf][sdst + 8] = rb1;                                      \
        } while (0)
    #define STAGE_DRAIN() do { } while (0)
#endif

    // Prologue: stage K-chunk 0 into buffer 0.
    STAGE_ISSUE(0, 0);
    STAGE_COMMIT(0);
    STAGE_DRAIN();
    __syncthreads();

#pragma unroll 1
    for (int s = 0; s < NSTEP; ++s) {
        const int cur = s & 1;

        if (s + 1 < NSTEP)
            STAGE_ISSUE(1 - cur, (s + 1) * 32);

        const _Float16* As = lsA[cur];
        const _Float16* Bs = lsB[cur];

        // A fragments (reused across the 4 N-tiles)
        v16h a[2];
#pragma unroll
        for (int mi = 0; mi < 2; ++mi) {
            const int la = (mg * 32 + mi * 16 + l16) * LDSTR;
            v8h lo = *(const v8h*)(As + la + lhalf * 8);
            v8h hi = *(const v8h*)(As + la + 16 + lhalf * 8);
            a[mi] = __builtin_shufflevector(lo, hi,
                        0,1,2,3,4,5,6,7,8,9,10,11,12,13,14,15);
        }

#pragma unroll
        for (int nt = 0; nt < 4; ++nt) {
            const int lb = (ng * 64 + nt * 16 + l16) * LDSTR + lhalf * 8;
            v8h blo = *(const v8h*)(Bs + lb);
            v8h bhi = *(const v8h*)(Bs + lb + 16);
            v16h b  = __builtin_shufflevector(blo, bhi,
                        0,1,2,3,4,5,6,7,8,9,10,11,12,13,14,15);

            acc[0][nt] = __builtin_amdgcn_wmma_f32_16x16x32_f16(
                             false, a[0], false, b, (short)0, acc[0][nt], false, false);
            acc[1][nt] = __builtin_amdgcn_wmma_f32_16x16x32_f16(
                             false, a[1], false, b, (short)0, acc[1][nt], false, false);
        }

        if (s + 1 < NSTEP)
            STAGE_COMMIT(1 - cur);
        STAGE_DRAIN();
        __syncthreads();
    }

    // Epilogue: D layout -> row = base + 8*lhalf + v, col = base + l16
#pragma unroll
    for (int mi = 0; mi < 2; ++mi) {
#pragma unroll
        for (int nt = 0; nt < 4; ++nt) {
            const int col  = nBase + ng * 64 + nt * 16 + l16;
            const float bb = b2[col];
            const int row0 = mBase + mg * 32 + mi * 16 + lhalf * 8;
#pragma unroll
            for (int v = 0; v < 8; ++v)
                out[(size_t)(row0 + v) * EMBED + col] = acc[mi][nt][v] + bb;
        }
    }
}

// ---------------------------------------------------------------------------
// Launch
// ---------------------------------------------------------------------------
extern "C" void kernel_launch(void* const* d_in, const int* in_sizes, int n_in,
                              void* d_out, int out_size, void* d_ws, size_t ws_size,
                              hipStream_t stream) {
    const float* x   = (const float*)d_in[0];   // (16,1024,512)
    const float* rxp = (const float*)d_in[1];   // (10,)
    const float* w1  = (const float*)d_in[2];   // (2048,10)
    const float* b1  = (const float*)d_in[3];   // (2048,)
    const float* w2  = (const float*)d_in[4];   // (512,2048)
    const float* b2  = (const float*)d_in[5];   // (512,)
    float* out = (float*)d_out;                 // (16,1024,512)

    // workspace: [0, 2MB) w2 in f16 ; [2MB, 2MB+64MB) h in f16
    _Float16* w2h = (_Float16*)d_ws;
    _Float16* h   = (_Float16*)((char*)d_ws + (2u << 20));

    w2_cvt_kernel<<<(512 * 2048) / 256, 256, 0, stream>>>(w2, w2h);
    qffn1_kernel<<<NTOK, 256, 0, stream>>>(x, rxp, w1, b1, h);
    qffn2_wmma_kernel<<<dim3(EMBED / BLK_N, NTOK / BLK_M), 256, 0, stream>>>(h, w2h, b2, out);
}
